// ContrastiveLoss_28372553957601
// MI455X (gfx1250) — compile-verified, matrix operand order checked by reading, not run on hardware
//
#include <hip/hip_runtime.h>
#include <hip/hip_bf16.h>

typedef __attribute__((ext_vector_type(16))) __bf16 v16bf;
typedef __attribute__((ext_vector_type(8)))  __bf16 v8bf;
typedef __attribute__((ext_vector_type(8)))  float  v8f;

static constexpr int N = 8192;
static constexpr int D = 512;

// ---- workspace layout (bytes) ----
static constexpr size_t SZ_FEAT = (size_t)N * D * 2;            // 8 MiB per bf16 plane
static constexpr size_t OFF_CNHI = 0;
static constexpr size_t OFF_CNLO = SZ_FEAT;
static constexpr size_t OFF_ENHI = 2 * SZ_FEAT;
static constexpr size_t OFF_ENLO = 3 * SZ_FEAT;
static constexpr size_t OFF_DIAG = 4 * SZ_FEAT;                  // N floats
static constexpr size_t OFF_ROWP = OFF_DIAG + (size_t)N * 4;     // 8  x N floats
static constexpr size_t OFF_COLP = OFF_ROWP + (size_t)8 * N * 4; // 128 x N floats

// ---- dynamic LDS layout for the GEMM kernel (bytes) ----
// ldsA: 64 rows x (hi 512 | lo 512 | pad 8) bf16 = 2064B/row -> 132096 B.
//       stride 2064B = 516 DW == 4 (mod 64 banks): conflict-free b128 reads.
// ldsB: 2 bufs x 64 rows x (hi 32 | lo 32 | pad 8) bf16 = 144B/row (36 DW,
//       36*ll mod 64 hits 16 distinct 4-bank groups): conflict-free.
static constexpr int      A_STRIDE   = 1032;              // elements per A row
static constexpr int      B_STRIDE   = 72;                // elements per B row
static constexpr uint32_t LDSA_BYTES = 64u * 2064u;       // 132096
static constexpr uint32_t LDSB_BUF   = 64u * 144u;        // 9216 per buffer
static constexpr uint32_t LDSB_OFF   = LDSA_BYTES;
static constexpr uint32_t COLACC_OFF = LDSB_OFF + 2 * LDSB_BUF;  // 150528
static constexpr uint32_t COLSH_OFF  = COLACC_OFF + 4096;        // [4][32] floats
static constexpr uint32_t ROWSH_OFF  = COLSH_OFF + 512;          // [2][64] floats
static constexpr uint32_t SMEM_BYTES = ROWSH_OFF + 512;          // 155648 B

// -------------------------------------------------------------------------
// Kernel 1: row L2-normalize (eps-clamped) + split into bf16 hi/lo planes.
// -------------------------------------------------------------------------
__global__ __launch_bounds__(256)
void cl_normalize_split(const float* __restrict__ cxr,
                        const float* __restrict__ ehr,
                        __bf16* __restrict__ cnHi, __bf16* __restrict__ cnLo,
                        __bf16* __restrict__ enHi, __bf16* __restrict__ enLo)
{
    const int row = blockIdx.x;
    const float* src;
    __bf16 *dHi, *dLo;
    if (row < N) {
        src = cxr + (size_t)row * D;
        dHi = cnHi + (size_t)row * D;  dLo = cnLo + (size_t)row * D;
    } else {
        const int r = row - N;
        src = ehr + (size_t)r * D;
        dHi = enHi + (size_t)r * D;    dLo = enLo + (size_t)r * D;
    }
    const int tid = threadIdx.x;
    const float x0 = src[tid];
    const float x1 = src[tid + 256];
    float ss = x0 * x0 + x1 * x1;
    #pragma unroll
    for (int m = 1; m < 32; m <<= 1) ss += __shfl_xor(ss, m, 32);

    __shared__ float wsum[8];
    __shared__ float sInv;
    const int wv = tid >> 5, ln = tid & 31;
    if (ln == 0) wsum[wv] = ss;
    __syncthreads();
    if (tid == 0) {
        float t = 0.f;
        #pragma unroll
        for (int i = 0; i < 8; ++i) t += wsum[i];
        float nrm = fmaxf(sqrtf(t), 1e-8f);   // matches jnp.maximum(n, EPS)
        sInv = 1.0f / nrm;
    }
    __syncthreads();
    const float inv = sInv;
    const float y0 = x0 * inv, y1 = x1 * inv;
    const __bf16 h0 = (__bf16)y0;  const __bf16 l0 = (__bf16)(y0 - (float)h0);
    const __bf16 h1 = (__bf16)y1;  const __bf16 l1 = (__bf16)(y1 - (float)h1);
    dHi[tid]       = h0;  dLo[tid]       = l0;
    dHi[tid + 256] = h1;  dLo[tid + 256] = l1;
}

// -------------------------------------------------------------------------
// WMMA bf16 fragment read. ISA layout for 16-bit operands of
// V_WMMA_F32_16X16X32_BF16: lane L -> row/col L%16; lanes 0-15 hold
// K {0..7, 16..23}, lanes 16-31 hold K {8..15, 24..31} of the 32-K slice
// starting at `base`. Two 16B contiguous chunks per lane.
// -------------------------------------------------------------------------
__device__ __forceinline__ v16bf cl_lds_frag(const __bf16* base, int laneHalf)
{
    union { v16bf v; v8bf h[2]; } u;
    u.h[0] = *(const v8bf*)(base + laneHalf * 8);
    u.h[1] = *(const v8bf*)(base + 16 + laneHalf * 8);
    return u.v;
}

__device__ __forceinline__ v8f cl_wmma(v16bf a, v16bf b, v8f c)
{
    return __builtin_amdgcn_wmma_f32_16x16x32_bf16(
        /*neg_a=*/false, a, /*neg_b=*/false, b,
        /*c_mod=*/(short)0, c, /*reuse_a=*/false, /*reuse_b=*/false);
}

// One per-lane 16B async copy global -> LDS (CDNA5, tracked by ASYNCcnt).
__device__ __forceinline__ void cl_async_b128(uint32_t lds_off, const __bf16* g)
{
    asm volatile("global_load_async_to_lds_b128 %0, %1, off"
                 :: "v"(lds_off), "v"(g) : "memory");
}

// Stage one 64B row-slice (4 x b128) of a B staging chunk.
__device__ __forceinline__ void cl_stage_chunk(uint32_t dstBase, const __bf16* src)
{
    #pragma unroll
    for (int i = 0; i < 4; ++i)
        cl_async_b128(dstBase + 16u * (uint32_t)i, src + 8 * i);
}

// -------------------------------------------------------------------------
// Kernel 2: fused split-bf16 WMMA GEMM + masked exp + deterministic partial
// row/col reductions.
//  - Block = 128 threads (4 waves) -> 64x64 output per column block,
//    each wave owns a 32x32 quadrant (8 fragments / 12 WMMAs per K-chunk).
//  - A strip (64 rows x 512 K, hi+lo) async-staged ONCE into padded LDS.
//  - B 64-col x 32-K chunks async-staged into padded LDS, double-buffered.
//  - Fixed lse shift M = 1/T (max |logit|), so plain exp-sums; the masked
//    diagonal contributes exp(NEG_FILL-M) == 0 and is simply skipped.
// -------------------------------------------------------------------------
__global__ __launch_bounds__(128)
void cl_gemm_lse(const __bf16* __restrict__ cnHi, const __bf16* __restrict__ cnLo,
                 const __bf16* __restrict__ enHi, const __bf16* __restrict__ enLo,
                 const float* __restrict__ temperature,
                 float* __restrict__ diagOut,
                 float* __restrict__ rowPartial,   // [8][N]
                 float* __restrict__ colPartial)   // [128][N]
{
    const int strip = blockIdx.x;     // 0..127 : rows strip*64 .. +63
    const int part  = blockIdx.y;     // 0..7   : cols part*1024 .. +1023
    const int tid   = threadIdx.x;
    const int w     = tid >> 5;       // wave 0..3 -> quadrant (w>>1, w&1)
    const int lane  = tid & 31;
    const int lh    = lane >> 4;      // lane half (K-group / +8 row select)
    const int ll    = lane & 15;      // row (A) / column (B,C) within tile
    const int wr    = (w >> 1) * 32;  // wave's row offset within strip
    const int wc    = (w & 1) * 32;   // wave's col offset within 64-block

    const float invT  = 1.0f / temperature[0];
    const float shift = invT;

    extern __shared__ __align__(16) char smem[];
    __bf16* ldsA  = (__bf16*)smem;                     // [64][hi512|lo512|pad8]
    __bf16* ldsB  = (__bf16*)(smem + LDSB_OFF);        // [2][64][hi32|lo32|pad8]
    float*  colAcc = (float*)(smem + COLACC_OFF);      // [1024]
    float (*colSh)[32] = (float(*)[32])(smem + COLSH_OFF); // [4][32]
    float (*rowSh)[64] = (float(*)[64])(smem + ROWSH_OFF); // [2][64]

    for (int c = tid; c < 1024; c += 128) colAcc[c] = 0.f;
    if (tid < 64) { rowSh[0][tid] = 0.f; rowSh[1][tid] = 0.f; }

    // ---- stage the whole A strip once (thread -> row, plane) ----
    {
        const int aRow = tid >> 1, aPlane = tid & 1;
        const __bf16* src = (aPlane ? cnLo : cnHi) + (size_t)(strip * 64 + aRow) * D;
        const uint32_t dst = (uint32_t)(uintptr_t)ldsA
                           + (uint32_t)(aRow * 2064 + aPlane * 1024);
        #pragma unroll 4
        for (int i = 0; i < 64; ++i)
            cl_async_b128(dst + 16u * (uint32_t)i, src + 8 * i);
    }

    // B staging assignment: thread -> (row in 64-block, plane hi/lo)
    const int sRow   = tid >> 1;
    const int sPlane = tid & 1;
    const uint32_t ldsBBase = (uint32_t)(uintptr_t)ldsB;
    const uint32_t stageDst = ldsBBase + (uint32_t)(sRow * 144 + sPlane * 64);
    const __bf16* planeSrc  = sPlane ? enLo : enHi;

    // per-lane A fragment bases (row stride A_STRIDE elems; lo plane at +512)
    const __bf16* aRow0 = ldsA + (size_t)(wr + ll) * A_STRIDE;
    const __bf16* aRow1 = aRow0 + 16 * A_STRIDE;

    for (int jb = 0; jb < 16; ++jb) {
        const int j0 = part * 1024 + jb * 64;
        const __bf16* stageSrcRow = planeSrc + (size_t)(j0 + sRow) * D;

        v8f acc00 = {}, acc01 = {}, acc10 = {}, acc11 = {};

        cl_stage_chunk(stageDst, stageSrcRow);            // chunk 0 -> buf 0

        for (int kc = 0; kc < 16; ++kc) {
            __syncthreads();   // prior readers of the buffer being refilled
            if (kc + 1 < 16) {
                cl_stage_chunk(stageDst + (uint32_t)(((kc + 1) & 1) * LDSB_BUF),
                               stageSrcRow + (kc + 1) * 32);
                asm volatile("s_wait_asynccnt 0x4" ::: "memory");
            } else {
                asm volatile("s_wait_asynccnt 0x0" ::: "memory");
            }
            __syncthreads();   // chunk kc (and, first time, all of A) visible

            const int k0 = kc * 32;
            const v16bf aHi0 = cl_lds_frag(aRow0 + k0, lh);
            const v16bf aLo0 = cl_lds_frag(aRow0 + 512 + k0, lh);
            const v16bf aHi1 = cl_lds_frag(aRow1 + k0, lh);
            const v16bf aLo1 = cl_lds_frag(aRow1 + 512 + k0, lh);

            const __bf16* bRow0 = ldsB + (kc & 1) * (LDSB_BUF / 2)
                                + (size_t)(wc + ll) * B_STRIDE;
            const __bf16* bRow1 = bRow0 + 16 * B_STRIDE;
            const v16bf bHi0 = cl_lds_frag(bRow0, lh);
            const v16bf bLo0 = cl_lds_frag(bRow0 + 32, lh);
            const v16bf bHi1 = cl_lds_frag(bRow1, lh);
            const v16bf bLo1 = cl_lds_frag(bRow1 + 32, lh);

            // bf16x3 split product: hi*hi + hi*lo + lo*hi per 16x16 tile
            acc00 = cl_wmma(aHi0, bHi0, acc00);
            acc00 = cl_wmma(aHi0, bLo0, acc00);
            acc00 = cl_wmma(aLo0, bHi0, acc00);

            acc01 = cl_wmma(aHi0, bHi1, acc01);
            acc01 = cl_wmma(aHi0, bLo1, acc01);
            acc01 = cl_wmma(aLo0, bHi1, acc01);

            acc10 = cl_wmma(aHi1, bHi0, acc10);
            acc10 = cl_wmma(aHi1, bLo0, acc10);
            acc10 = cl_wmma(aLo1, bHi0, acc10);

            acc11 = cl_wmma(aHi1, bHi1, acc11);
            acc11 = cl_wmma(aHi1, bLo1, acc11);
            acc11 = cl_wmma(aLo1, bHi1, acc11);
        }

        // ---- epilogue: C layout lane = column ll, rows M = v + 8*lh ----
        float rowC[2][8];
        #pragma unroll
        for (int tr = 0; tr < 2; ++tr)
            #pragma unroll
            for (int v = 0; v < 8; ++v) rowC[tr][v] = 0.f;
        float csv[2] = {0.f, 0.f};

        #define CL_EPI(ACC, TR, TC)                                          \
        {                                                                    \
            _Pragma("unroll")                                                \
            for (int v = 0; v < 8; ++v) {                                    \
                const float x  = ACC[v] * invT;                              \
                const int   gi = strip * 64 + wr + (TR) * 16 + v + 8 * lh;   \
                const int   gj = j0 + wc + (TC) * 16 + ll;                   \
                float e;                                                     \
                if (gi == gj) { diagOut[gi] = x; e = 0.f; }                  \
                else          { e = __expf(x - shift); }                     \
                rowC[TR][v] += e;                                            \
                csv[TC]     += e;                                            \
            }                                                                \
        }
        CL_EPI(acc00, 0, 0)
        CL_EPI(acc01, 0, 1)
        CL_EPI(acc10, 1, 0)
        CL_EPI(acc11, 1, 1)
        #undef CL_EPI

        // row sums: reduce across the 16 lanes (columns) of each half
        #pragma unroll
        for (int tr = 0; tr < 2; ++tr)
            #pragma unroll
            for (int v = 0; v < 8; ++v) {
                float r = rowC[tr][v];
                r += __shfl_xor(r, 1, 32);
                r += __shfl_xor(r, 2, 32);
                r += __shfl_xor(r, 4, 32);
                r += __shfl_xor(r, 8, 32);
                if (ll == 0)
                    rowSh[w & 1][wr + tr * 16 + v + 8 * lh] += r; // unique owner
            }

        // col sums: fold the two 8-row halves, publish per-wave 32 cols
        csv[0] += __shfl_xor(csv[0], 16, 32);
        csv[1] += __shfl_xor(csv[1], 16, 32);
        if (lane < 16) {
            colSh[w][ 0 + lane] = csv[0];
            colSh[w][16 + lane] = csv[1];
        }
        __syncthreads();
        if (tid < 64) {
            const int half = tid >> 5, cl = tid & 31;
            colAcc[jb * 64 + tid] += colSh[half][cl] + colSh[half + 2][cl];
        }
        __syncthreads();
    }

    // flush disjoint partials (unique writers -> deterministic)
    if (tid < 64)
        rowPartial[(size_t)part * N + strip * 64 + tid]
            = rowSh[0][tid] + rowSh[1][tid];
    for (int c = tid; c < 1024; c += 128)
        colPartial[(size_t)strip * N + part * 1024 + c] = colAcc[c];
}

// -------------------------------------------------------------------------
// Kernel 3: combine partials in fixed order, compute scalar loss.
// loss = (1/N) * sum_i [ lse_row_i + lse_col_i - 2*diag_i ]
// -------------------------------------------------------------------------
__global__ __launch_bounds__(256)
void cl_finalize(const float* __restrict__ diag,
                 const float* __restrict__ rowPartial,
                 const float* __restrict__ colPartial,
                 const float* __restrict__ temperature,
                 float* __restrict__ out)
{
    const float shift = 1.0f / temperature[0];
    float local = 0.f;
    for (int i = threadIdx.x; i < N; i += 256) {
        float rs = 0.f;
        #pragma unroll
        for (int y = 0; y < 8; ++y) rs += rowPartial[(size_t)y * N + i];
        float csum = 0.f;
        for (int s = 0; s < 128; ++s) csum += colPartial[(size_t)s * N + i];
        local += (shift + logf(rs)) + (shift + logf(csum)) - 2.0f * diag[i];
    }
    #pragma unroll
    for (int m = 1; m < 32; m <<= 1) local += __shfl_xor(local, m, 32);
    __shared__ float wsum[8];
    const int wv = threadIdx.x >> 5, ln = threadIdx.x & 31;
    if (ln == 0) wsum[wv] = local;
    __syncthreads();
    if (threadIdx.x == 0) {
        float tot = 0.f;
        #pragma unroll
        for (int i = 0; i < 8; ++i) tot += wsum[i];
        out[0] = tot / (float)N;
    }
}

// -------------------------------------------------------------------------
extern "C" void kernel_launch(void* const* d_in, const int* in_sizes, int n_in,
                              void* d_out, int out_size, void* d_ws, size_t ws_size,
                              hipStream_t stream)
{
    (void)in_sizes; (void)n_in; (void)out_size; (void)ws_size;
    const float* cxr  = (const float*)d_in[0];
    const float* ehr  = (const float*)d_in[1];
    const float* temp = (const float*)d_in[2];
    float* out = (float*)d_out;

    char* ws = (char*)d_ws;
    __bf16* cnHi = (__bf16*)(ws + OFF_CNHI);
    __bf16* cnLo = (__bf16*)(ws + OFF_CNLO);
    __bf16* enHi = (__bf16*)(ws + OFF_ENHI);
    __bf16* enLo = (__bf16*)(ws + OFF_ENLO);
    float*  diag = (float*)(ws + OFF_DIAG);
    float*  rowP = (float*)(ws + OFF_ROWP);
    float*  colP = (float*)(ws + OFF_COLP);

    cl_normalize_split<<<2 * N, 256, 0, stream>>>(cxr, ehr, cnHi, cnLo, enHi, enLo);
    cl_gemm_lse<<<dim3(N / 64, 8), 128, SMEM_BYTES, stream>>>(cnHi, cnLo, enHi, enLo,
                                                              temp, diag, rowP, colP);
    cl_finalize<<<1, 256, 0, stream>>>(diag, rowP, colP, temp, out);
}
